// SGAE_82858509075197
// MI455X (gfx1250) — matrix operand order repaired and checked by reading, not compiled, not run on hardware
//
#include <hip/hip_runtime.h>
#include <hip/hip_bf16.h>
#include <stdint.h>

// ---------------------------------------------------------------------------
// Spiking graph autoencoder forward pass for MI455X (gfx1250, wave32, WMMA).
//  * Dense layers: v_wmma_f32_16x16x32_f16 (spikes are exact 0/1 -> f16 A is
//    lossless; f32 accumulate feeds the fused LIF epilogue).
//  * Weight (B) fragments are pre-packed to per-lane-contiguous layout and
//    staged block-wide into LDS by the Tensor Data Mover (tensor_load_to_lds,
//    TENSORcnt). A runtime-selectable cooperative-copy fallback (host passes
//    0) keeps the LDS stores visible to the compiler so the fragment reads
//    lower to real ds_load_b128 (otherwise LLVM folds loads from a
//    never-stored __shared__ array to undef).
//  * One wave = 1 row tile x 4 (or 2) col tiles -> A fragment reuse.
//  * Edge propagate gathers f16 features (L2-resident: tables are 25-51MB vs
//    192MB L2) and scatter-adds with f32 global atomics.
// ---------------------------------------------------------------------------

#define N_NODESD 50000
#define N_EDGESD 400000
#define N_LINKSD 50000
#define N_FEATD  256
#define ENC0D    256
#define ENC1D    128
#define DECD     64
#define CATD     (N_FEATD + ENC0D)   // 512
#define TSTEPS   4
#define THRV     0.2f
#define DECAYV   0.5f                // 1 - 1/TAU, TAU = 2

typedef __attribute__((ext_vector_type(16))) _Float16 v16h;
typedef __attribute__((ext_vector_type(8)))  _Float16 v8h;
typedef __attribute__((ext_vector_type(8)))  float    v8f;
typedef unsigned int u32x4 __attribute__((ext_vector_type(4)));
typedef int          i32x8 __attribute__((ext_vector_type(8)));
typedef int          i32x4 __attribute__((ext_vector_type(4)));

// --------------------------- utilities -------------------------------------

__device__ __forceinline__ float hash_u01(uint32_t a, uint32_t b) {
  uint32_t h = a * 0x9E3779B9u + b * 0x85EBCA6Bu + 0x6A09E667u;
  h ^= h >> 16; h *= 0x7FEB352Du;
  h ^= h >> 15; h *= 0x846CA68Bu;
  h ^= h >> 16;
  return (float)(h >> 8) * (1.0f / 16777216.0f);
}

// A-fragment load per CDNA5 16-bit A layout (16x32 MxK): lanes 0-15 (half=0)
// hold K = {0..7,16..23}; lanes 16-31 hold K = {8..15,24..31}, as dword pairs.
__device__ __forceinline__ v16h load_a_frag(const _Float16* Arow, int half) {
  const uint32_t* Ar = (const uint32_t*)Arow;   // dword = 2 halves
  union { v16h h; uint32_t u[8]; } a;
  const int o = half * 4;
#pragma unroll
  for (int j = 0; j < 4; ++j) a.u[j]     = Ar[o + j];
#pragma unroll
  for (int j = 0; j < 4; ++j) a.u[4 + j] = Ar[o + 8 + j];
  return a.h;
}

__device__ __forceinline__ v8f wmma_f16(v16h a, v16h b, v8f c) {
  return __builtin_amdgcn_wmma_f32_16x16x32_f16(false, a, false, b,
                                                (short)0, c, false, false);
}

// ---------------------- Tensor Data Mover B staging -------------------------
// 2D D# (ISA 8.3/8.4): copy `rows` rows of `elems` f16 each, source row pitch
// `row_stride_elems` (f16 units), packed densely into LDS at lds_byte_off.
// This toolchain's builtin takes 6 args (g0, g1, g2, g3, g4, cpol).
__device__ __forceinline__ void tdm_load_2d_f16(const _Float16* gsrc,
                                                uint32_t lds_byte_off,
                                                uint32_t elems, uint32_t rows,
                                                uint32_t row_stride_elems) {
  const uint64_t ga = (uint64_t)(uintptr_t)gsrc;
  u32x4 g0;
  g0[0] = 1u;                                      // count=1, user desc
  g0[1] = lds_byte_off;                            // lds_addr (bytes)
  g0[2] = (uint32_t)(ga & 0xFFFFFFFFu);            // global_addr[31:0]
  g0[3] = (uint32_t)((ga >> 32) & 0x01FFFFFFu)     // global_addr[56:32]
        | (2u << 30);                              // type = 2 ("image")
  i32x8 g1;
  g1[0] = (int)(1u << 16);                         // wg_mask=0, data_size=1 (2B)
  g1[1] = (int)((elems & 0xFFFFu) << 16);          // tensor_dim0[15:0]
  g1[2] = (int)((elems >> 16) | ((rows & 0xFFFFu) << 16));   // dim0 hi | dim1 lo
  g1[3] = (int)((rows >> 16) | ((elems & 0xFFFFu) << 16));   // dim1 hi | tile_dim0
  g1[4] = (int)(rows & 0xFFFFu);                   // tile_dim1 | tile_dim2=0
  g1[5] = (int)row_stride_elems;                   // tensor_dim0_stride[31:0]
  g1[6] = 0;                                       // stride hi | dim1_stride lo
  g1[7] = 0;
  i32x4 z4 = {0, 0, 0, 0};
  i32x8 z8 = {0, 0, 0, 0, 0, 0, 0, 0};
  __builtin_amdgcn_tensor_load_to_lds(g0, g1, z4, z4, z8, 0);
}

// Stage packed B into LDS: TDM path (fb==0) or cooperative-copy fallback.
// The fallback keeps LDS stores visible to the compiler so sB reads are not
// folded to undef; host always passes fb = 0.
__device__ __forceinline__ void stage_b(int fb, _Float16* sB,
                                        const _Float16* gsrc, uint32_t elems,
                                        uint32_t rows, uint32_t stride) {
  if (fb) {
    const int total = (int)(elems * rows);
    for (int i = threadIdx.x * 8; i < total; i += blockDim.x * 8) {
      const int r = i / (int)elems, jc = i % (int)elems;
      *(v8h*)(sB + i) = *(const v8h*)(gsrc + (size_t)r * stride + jc);
    }
  } else if ((threadIdx.x >> 5) == 0) {
    tdm_load_2d_f16(gsrc, 0, elems, rows, stride);
    __builtin_amdgcn_s_wait_tensorcnt(0);
  }
  __syncthreads();
}

// --------------------------- weight packing --------------------------------
// Bp[((kt*NT + nt)*32 + lane)*16 + i] = B[kt*32 + (lane>>4)*8 + (i&7) + (i>>3)*16,
//                                         nt*16 + (lane&15)]  (f32 -> f16)
__global__ void k_pack_b(const float* __restrict__ B, _Float16* __restrict__ Bp,
                         int K, int Ncols) {
  const int NT = Ncols >> 4;
  const int total = (K >> 5) * NT * 512;
  for (int idx = blockIdx.x * blockDim.x + threadIdx.x; idx < total;
       idx += gridDim.x * blockDim.x) {
    const int i    = idx & 15;
    const int lane = (idx >> 4) & 31;
    const int tile = idx >> 9;
    const int nt = tile % NT, kt = tile / NT;
    const int n = (nt << 4) + (lane & 15);
    const int k = (kt << 5) + ((lane >> 4) << 3) + (i & 7) + ((i >> 3) << 4);
    Bp[idx] = (_Float16)B[(size_t)k * Ncols + n];
  }
}

// --------------------------- poisson sampling ------------------------------

__global__ void k_poisson_x(const float* __restrict__ x, _Float16* __restrict__ inp,
                            int n, uint32_t seed) {
  for (int i = blockIdx.x * blockDim.x + threadIdx.x; i < n;
       i += gridDim.x * blockDim.x) {
    const float u = hash_u01(seed, (uint32_t)i);
    inp[i] = (_Float16)((u <= x[i]) ? 1.0f : 0.0f);
  }
}

__global__ void k_poisson_h0(const float* __restrict__ h0, _Float16* __restrict__ hs0,
                             _Float16* __restrict__ inp1, uint32_t seed) {
  const int n = N_NODESD * N_FEATD;
  for (int i = blockIdx.x * blockDim.x + threadIdx.x; i < n;
       i += gridDim.x * blockDim.x) {
    const int row = i >> 8, col = i & 255;
    const float u = hash_u01(seed, (uint32_t)i);
    const float s = (u <= h0[i]) ? 1.0f : 0.0f;
    hs0[i] = (_Float16)s;
    inp1[(size_t)row * CATD + N_FEATD + col] = (_Float16)s;   // concat right half
  }
}

// --------------------------- graph propagate -------------------------------
// h[dst] += w_e * feat[src]; 16B f16 gathers (L2-resident), f32 atomic scatter.
__global__ void k_prop(const _Float16* __restrict__ feat, const int* __restrict__ esrc,
                       const int* __restrict__ edst, const float* __restrict__ ew,
                       float* __restrict__ acc, int F) {
  const int per_edge = F >> 3;                       // 8 feats / thread
  const size_t total = (size_t)N_EDGESD * per_edge;
  for (size_t g = (size_t)blockIdx.x * blockDim.x + threadIdx.x; g < total;
       g += (size_t)gridDim.x * blockDim.x) {
    const int e = (int)(g / per_edge);
    const int f = (int)(g % per_edge) << 3;
    if ((g & 63) == 0 && e + 2048 < N_EDGESD) {
      __builtin_prefetch(esrc + e + 2048, 0, 1);     // edge stream lookahead
      __builtin_prefetch(ew + e + 2048, 0, 1);
    }
    const int s = esrc[e], d = edst[e];
    const float w = ew[e];
    const v8h fp = *(const v8h*)(feat + (size_t)s * F + f);
    float* ap = acc + (size_t)d * F + f;
#pragma unroll
    for (int j = 0; j < 8; ++j) atomicAdd(ap + j, w * (float)fp[j]);
  }
}

// --------------------------- elementwise LIF (layer-1 hidden) --------------
__global__ void k_lif_h1(const float* __restrict__ h1, float* __restrict__ v1h,
                         _Float16* __restrict__ hs1) {
  const int n = N_NODESD * CATD;
  for (int i = blockIdx.x * blockDim.x + threadIdx.x; i < n;
       i += gridDim.x * blockDim.x) {
    float v = v1h[i] * DECAYV + h1[i];
    const float s = (v - THRV >= 0.0f) ? 1.0f : 0.0f;
    v1h[i] = v - s * THRV;
    hs1[i] = (_Float16)s;
  }
}

// --------------------------- WMMA GEMM + fused LIF -------------------------
// 256-thread block = 8 waves = 8 row tiles; blockIdx.y = column-tile group.
// Wave 0 TDM-stages the packed B fragments for the whole group into LDS.

// Layer 1: (N x 256) @ (256 x 256), col groups of 4 -> LIF(v0z) -> zs0 (inp1).
__global__ __launch_bounds__(256)
void k_gemm_lif1(const _Float16* __restrict__ A, const _Float16* __restrict__ Bp,
                 float* __restrict__ v0z, _Float16* __restrict__ inp1, int fb) {
  __shared__ _Float16 sB[(N_FEATD / 32) * 4 * 512];          // 32KB
  const int lane = threadIdx.x & 31;
  const int wave = threadIdx.x >> 5;
  const int half = lane >> 4;
  const int mn   = lane & 15;
  const int NT   = ENC0D / 16;                               // 16
  const int group = blockIdx.y;                              // 4 groups of 4
  stage_b(fb, sB, Bp + (size_t)group * 4 * 512, 4 * 512, N_FEATD / 32, NT * 512);
  const int tm = blockIdx.x * 8 + wave;
  if (tm < N_NODESD / 16) {
    v8f c[4] = {};
    const _Float16* Arow = A + (size_t)(tm * 16 + mn) * N_FEATD;
#pragma unroll
    for (int kt = 0; kt < N_FEATD / 32; ++kt) {
      v16h a = load_a_frag(Arow + kt * 32, half);
      const _Float16* bb = sB + (size_t)kt * 4 * 512 + lane * 16;
#pragma unroll
      for (int g = 0; g < 4; ++g)
        c[g] = wmma_f16(a, *(const v16h*)(bb + g * 512), c[g]);
    }
#pragma unroll
    for (int g = 0; g < 4; ++g) {
      const int tn = group * 4 + g;
#pragma unroll
      for (int r = 0; r < 8; ++r) {
        const int row = tm * 16 + r + 8 * half;              // M = r + 8*half
        const int col = tn * 16 + mn;
        const size_t vi = (size_t)row * ENC0D + col;
        float v = v0z[vi] * DECAYV + c[g][r];
        const float s = (v - THRV >= 0.0f) ? 1.0f : 0.0f;
        v0z[vi] = v - s * THRV;
        inp1[(size_t)row * CATD + col] = (_Float16)s;        // concat left half
      }
    }
  }
}

// Layer 2: (N x 512) @ (512 x 128), col groups of 2 -> LIF(v1z) -> zs1 (f16).
__global__ __launch_bounds__(256)
void k_gemm_lif2(const _Float16* __restrict__ A, const _Float16* __restrict__ Bp,
                 float* __restrict__ v1z, _Float16* __restrict__ zs1, int fb) {
  __shared__ _Float16 sB[(CATD / 32) * 2 * 512];             // 32KB
  const int lane = threadIdx.x & 31;
  const int wave = threadIdx.x >> 5;
  const int half = lane >> 4;
  const int mn   = lane & 15;
  const int NT   = ENC1D / 16;                               // 8
  const int group = blockIdx.y;                              // 4 groups of 2
  stage_b(fb, sB, Bp + (size_t)group * 2 * 512, 2 * 512, CATD / 32, NT * 512);
  const int tm = blockIdx.x * 8 + wave;
  if (tm < N_NODESD / 16) {
    v8f c[2] = {};
    const _Float16* Arow = A + (size_t)(tm * 16 + mn) * CATD;
#pragma unroll
    for (int kt = 0; kt < CATD / 32; ++kt) {
      v16h a = load_a_frag(Arow + kt * 32, half);
      const _Float16* bb = sB + (size_t)kt * 2 * 512 + lane * 16;
#pragma unroll
      for (int g = 0; g < 2; ++g)
        c[g] = wmma_f16(a, *(const v16h*)(bb + g * 512), c[g]);
    }
#pragma unroll
    for (int g = 0; g < 2; ++g) {
      const int tn = group * 2 + g;
#pragma unroll
      for (int r = 0; r < 8; ++r) {
        const int row = tm * 16 + r + 8 * half;
        const int col = tn * 16 + mn;
        const size_t vi = (size_t)row * ENC1D + col;
        float v = v1z[vi] * DECAYV + c[g][r];
        const float s = (v - THRV >= 0.0f) ? 1.0f : 0.0f;
        v1z[vi] = v - s * THRV;
        zs1[vi] = (_Float16)s;
      }
    }
  }
}

// Decoder: gathered (2L x 128) @ (128 x 64) + bd -> LIF(vdec) -> zd (f32).
__global__ __launch_bounds__(256)
void k_dec_gemm(const _Float16* __restrict__ zs1, const _Float16* __restrict__ Bp,
                const float* __restrict__ bd, const int* __restrict__ nodes_s,
                const int* __restrict__ nodes_t, float* __restrict__ vdec,
                float* __restrict__ zd, int fb) {
  __shared__ _Float16 sB[(ENC1D / 32) * 4 * 512];            // 16KB (all of Wd)
  const int lane = threadIdx.x & 31;
  const int wave = threadIdx.x >> 5;
  const int half = lane >> 4;
  const int mn   = lane & 15;
  const uint32_t LEN = (ENC1D / 32) * 4 * 512;               // contiguous
  stage_b(fb, sB, Bp, LEN, 1, LEN);
  const int tm = blockIdx.x * 8 + wave;
  if (tm < 2 * N_LINKSD / 16) {
    const int hr = tm * 16 + mn;                             // hcat row -> node
    const int node = (hr < N_LINKSD) ? nodes_s[hr] : nodes_t[hr - N_LINKSD];
    const _Float16* Arow = zs1 + (size_t)node * ENC1D;
    v8f c[4] = {};
#pragma unroll
    for (int kt = 0; kt < ENC1D / 32; ++kt) {
      v16h a = load_a_frag(Arow + kt * 32, half);
      const _Float16* bb = sB + (size_t)kt * 4 * 512 + lane * 16;
#pragma unroll
      for (int g = 0; g < 4; ++g)
        c[g] = wmma_f16(a, *(const v16h*)(bb + g * 512), c[g]);
    }
#pragma unroll
    for (int g = 0; g < 4; ++g) {
#pragma unroll
      for (int r = 0; r < 8; ++r) {
        const int row = tm * 16 + r + 8 * half;
        const int col = g * 16 + mn;
        const size_t vi = (size_t)row * DECD + col;
        float v = vdec[vi] * DECAYV + (c[g][r] + bd[col]);
        const float s = (v - THRV >= 0.0f) ? 1.0f : 0.0f;
        vdec[vi] = v - s * THRV;
        zd[vi] = s;
      }
    }
  }
}

// --------------------------- readout ---------------------------------------
__global__ void k_y_accum(const float* __restrict__ zd, const float* __restrict__ w_rec,
                          float* __restrict__ yacc) {
  const int l = blockIdx.x * blockDim.x + threadIdx.x;
  if (l >= N_LINKSD) return;
  const float* a = zd + (size_t)l * DECD;
  const float* b = zd + (size_t)(N_LINKSD + l) * DECD;
  float s = 0.0f;
#pragma unroll
  for (int k = 0; k < DECD; ++k) s += a[k] * w_rec[k] * b[k];
  yacc[l] += s;
}

__global__ void k_finalize(const float* __restrict__ yacc, float* __restrict__ out) {
  const int l = blockIdx.x * blockDim.x + threadIdx.x;
  if (l >= N_LINKSD) return;
  const float m = yacc[l] * (1.0f / (float)TSTEPS);
  out[l] = 1.0f / (1.0f + __expf(-m));
}

// --------------------------- host driver -----------------------------------

extern "C" void kernel_launch(void* const* d_in, const int* in_sizes, int n_in,
                              void* d_out, int out_size, void* d_ws, size_t ws_size,
                              hipStream_t stream) {
  const float* x        = (const float*)d_in[0];
  const float* W1       = (const float*)d_in[1];
  const float* W2       = (const float*)d_in[2];
  const float* Wd       = (const float*)d_in[3];
  const float* bd       = (const float*)d_in[4];
  const float* w_rec    = (const float*)d_in[5];
  const float* edge_w   = (const float*)d_in[6];
  const int*   edge_src = (const int*)d_in[7];
  const int*   edge_dst = (const int*)d_in[8];
  const int*   nodes_s  = (const int*)d_in[9];
  const int*   nodes_t  = (const int*)d_in[10];
  float* out = (float*)d_out;

  char* base = (char*)d_ws;
  size_t off = 0;
  auto carve = [&](size_t bytes) -> char* {
    char* p = base + off;
    off += (bytes + 255) & ~(size_t)255;
    return p;
  };
  _Float16* W1p  = (_Float16*)carve((size_t)N_FEATD * ENC0D * 2);
  _Float16* W2p  = (_Float16*)carve((size_t)CATD * ENC1D * 2);
  _Float16* Wdp  = (_Float16*)carve((size_t)ENC1D * DECD * 2);
  _Float16* inp  = (_Float16*)carve((size_t)N_NODESD * N_FEATD * 2);
  float*    h0   = (float*)   carve((size_t)N_NODESD * N_FEATD * 4);
  _Float16* hs0  = (_Float16*)carve((size_t)N_NODESD * N_FEATD * 2);
  _Float16* inp1 = (_Float16*)carve((size_t)N_NODESD * CATD * 2);
  float*    v0z  = (float*)   carve((size_t)N_NODESD * ENC0D * 4);
  float*    h1   = (float*)   carve((size_t)N_NODESD * CATD * 4);
  float*    v1h  = (float*)   carve((size_t)N_NODESD * CATD * 4);
  _Float16* hs1  = (_Float16*)carve((size_t)N_NODESD * CATD * 2);
  float*    v1z  = (float*)   carve((size_t)N_NODESD * ENC1D * 4);
  _Float16* zs1  = (_Float16*)carve((size_t)N_NODESD * ENC1D * 2);
  float*    vdec = (float*)   carve((size_t)2 * N_LINKSD * DECD * 4);
  float*    zd   = (float*)   carve((size_t)2 * N_LINKSD * DECD * 4);
  float*    yacc = (float*)   carve((size_t)N_LINKSD * 4);

  const int TB = 256;
  const int FB = 0;   // 0 = TDM staging path; 1 = cooperative-copy fallback

  k_pack_b<<<256, TB, 0, stream>>>(W1, W1p, N_FEATD, ENC0D);
  k_pack_b<<<256, TB, 0, stream>>>(W2, W2p, CATD, ENC1D);
  k_pack_b<<<64,  TB, 0, stream>>>(Wd, Wdp, ENC1D, DECD);

  (void)hipMemsetAsync(v0z,  0, (size_t)N_NODESD * ENC0D * 4, stream);
  (void)hipMemsetAsync(v1h,  0, (size_t)N_NODESD * CATD * 4, stream);
  (void)hipMemsetAsync(v1z,  0, (size_t)N_NODESD * ENC1D * 4, stream);
  (void)hipMemsetAsync(vdec, 0, (size_t)2 * N_LINKSD * DECD * 4, stream);
  (void)hipMemsetAsync(yacc, 0, (size_t)N_LINKSD * 4, stream);

  for (int t = 0; t < TSTEPS; ++t) {
    const uint32_t seed1 = 0x5EED0000u + 2u * (uint32_t)t;
    const uint32_t seed2 = seed1 + 1u;

    k_poisson_x<<<4096, TB, 0, stream>>>(x, inp, N_NODESD * N_FEATD, seed1);

    (void)hipMemsetAsync(h0, 0, (size_t)N_NODESD * N_FEATD * 4, stream);
    k_prop<<<8192, TB, 0, stream>>>(inp, edge_src, edge_dst, edge_w, h0, N_FEATD);

    k_poisson_h0<<<4096, TB, 0, stream>>>(h0, hs0, inp1, seed2);

    k_gemm_lif1<<<dim3(391, 4), TB, 0, stream>>>(hs0, W1p, v0z, inp1, FB);

    (void)hipMemsetAsync(h1, 0, (size_t)N_NODESD * CATD * 4, stream);
    k_prop<<<8192, TB, 0, stream>>>(inp1, edge_src, edge_dst, edge_w, h1, CATD);

    k_lif_h1<<<4096, TB, 0, stream>>>(h1, v1h, hs1);

    k_gemm_lif2<<<dim3(391, 4), TB, 0, stream>>>(hs1, W2p, v1z, zs1, FB);

    k_dec_gemm<<<dim3(782, 1), TB, 0, stream>>>(zs1, Wdp, bd, nodes_s, nodes_t,
                                                vdec, zd, FB);

    k_y_accum<<<(N_LINKSD + TB - 1) / TB, TB, 0, stream>>>(zd, w_rec, yacc);
  }

  k_finalize<<<(N_LINKSD + TB - 1) / TB, TB, 0, stream>>>(yacc, out);
}
